// MoeForward_81252191306060
// MI455X (gfx1250) — compile-verified, matrix operand order checked by reading, not run on hardware
//
#include <hip/hip_runtime.h>
#include <hip/hip_bf16.h>

// ---- problem constants (from reference setup_inputs) ----
#define T_TOK 4096   // B*S
#define D_DIM 1024
#define E_NUM 8
#define F_DIM 2048
// top_k = 2 (slot id = 2*t + rank)

typedef __attribute__((ext_vector_type(16))) __bf16 v16bf;
typedef __attribute__((ext_vector_type(8)))  float  v8f;

union FragB {
    v16bf v;
    uint4 q[2];
    unsigned u[8];
};

// pack two f32 -> one dword of 2x bf16 (round-to-nearest-even-ish)
__device__ __forceinline__ unsigned pk_bf16(float a, float b) {
    unsigned ua = __float_as_uint(a);
    unsigned ub = __float_as_uint(b);
    ua = (ua + 0x7FFFu + ((ua >> 16) & 1u)) >> 16;
    ub = (ub + 0x7FFFu + ((ub >> 16) & 1u)) >> 16;
    return ua | (ub << 16);
}

__device__ __forceinline__ unsigned short bf16_1(float a) {
    unsigned ua = __float_as_uint(a);
    return (unsigned short)((ua + 0x7FFFu + ((ua >> 16) & 1u)) >> 16);
}

// A/B fragment from a row-major bf16 matrix; caller supplies the (gathered) row base.
// 16-bit 16x32 layout: lanes 0-15 hold K {k0..k0+7, k0+16..k0+23}; lanes 16-31 hold
// K {k0+8..k0+15, k0+24..k0+31} (ISA 7.12.2) -> two 16B chunks per lane.
__device__ __forceinline__ v16bf frag_bf16_rows(const unsigned short* __restrict__ row,
                                                int k0, int lane) {
    const unsigned short* q = row + k0 + ((lane >> 4) << 3);
    FragB f;
    f.q[0] = *(const uint4*)(q);
    f.q[1] = *(const uint4*)(q + 16);
    return f.v;
}

// ---------------- kernel 0: zero output + expert counters ----------------
__global__ __launch_bounds__(256) void k_zero(float* __restrict__ out,
                                              int* __restrict__ counts) {
    size_t i = ((size_t)blockIdx.x * 256 + threadIdx.x) * 4;
    *(float4*)(out + i) = make_float4(0.f, 0.f, 0.f, 0.f);
    if (blockIdx.x == 0 && threadIdx.x < E_NUM) counts[threadIdx.x] = 0;
}

// ---------------- kernel 1: fp32 -> bf16 bulk convert (x and weights) --------
__global__ __launch_bounds__(256) void k_cvt(const float* __restrict__ src,
                                             unsigned short* __restrict__ dst) {
    size_t i = ((size_t)blockIdx.x * 256 + threadIdx.x) * 4;
    float4 v = *(const float4*)(src + i);
    uint2 o;
    o.x = pk_bf16(v.x, v.y);
    o.y = pk_bf16(v.z, v.w);
    *(uint2*)(dst + i) = o;
}

// ---------------- kernel 2: router + top-2 compaction ----------------
__global__ __launch_bounds__(256) void k_router(const float* __restrict__ x,
                                                const float* __restrict__ rw,
                                                int* __restrict__ counts,
                                                int* __restrict__ list,
                                                float* __restrict__ wlist) {
    int t = blockIdx.x * 256 + threadIdx.x;
    const float4* xp = (const float4*)(x + (size_t)t * D_DIM);
    float acc[E_NUM];
#pragma unroll
    for (int e = 0; e < E_NUM; ++e) acc[e] = 0.f;
    for (int d4 = 0; d4 < D_DIM / 4; ++d4) {
        float4 xv = xp[d4];
#pragma unroll
        for (int e = 0; e < E_NUM; ++e) {
            float4 wv = *(const float4*)(rw + (size_t)e * D_DIM + d4 * 4);
            acc[e] += xv.x * wv.x + xv.y * wv.y + xv.z * wv.z + xv.w * wv.w;
        }
    }
    // softmax (denominator cancels after top-2 renormalization)
    float mx = acc[0];
#pragma unroll
    for (int e = 1; e < E_NUM; ++e) mx = fmaxf(mx, acc[e]);
    float p[E_NUM];
#pragma unroll
    for (int e = 0; e < E_NUM; ++e) p[e] = __expf(acc[e] - mx);
    int i1 = 0; float v1 = p[0];
#pragma unroll
    for (int e = 1; e < E_NUM; ++e) if (p[e] > v1) { v1 = p[e]; i1 = e; }
    int i2 = (i1 == 0) ? 1 : 0; float v2 = p[i2];
#pragma unroll
    for (int e = 0; e < E_NUM; ++e)
        if (e != i1 && p[e] > v2) { v2 = p[e]; i2 = e; }
    float inv = 1.0f / (v1 + v2);
    int p1 = atomicAdd(&counts[i1], 1);
    list[i1 * T_TOK + p1]  = (t << 1);
    wlist[i1 * T_TOK + p1] = v1 * inv;
    int p2 = atomicAdd(&counts[i2], 1);
    list[i2 * T_TOK + p2]  = (t << 1) | 1;
    wlist[i2 * T_TOK + p2] = v2 * inv;
}

// ---------------- kernel 3: grouped gate/up GEMM + SiLU fuse ----------------
// grid = (F/64, T/64, E); block = 256 (8 waves).
// Block tile: 64 tokens x 64 F-cols for both gate and up.
// Wave w: nj = w&3 selects the 16-col tile; mg = w>>2 selects a pair of 16-row
// tiles; each wave runs 4 WMMAs per K-step (2 M-tiles x {gate,up}) off
// 2 A-fragments + 2 B-fragments (all pure bf16 b128 loads, zero conversion).
__global__ __launch_bounds__(256) void k_gateup(const unsigned short* __restrict__ xb,
                                                const unsigned short* __restrict__ gwb,
                                                const unsigned short* __restrict__ uwb,
                                                const int* __restrict__ counts,
                                                const int* __restrict__ list,
                                                unsigned short* __restrict__ hbuf) {
    const int e = blockIdx.z;
    const int cnt = counts[e];
    const int m0 = blockIdx.y * 64;
    if (m0 >= cnt) return;
    const int f0 = blockIdx.x * 64;
    const int lane = threadIdx.x & 31;
    const int wv = threadIdx.x >> 5;
    const int nj = wv & 3;      // 0..3 : 16-col tile
    const int mg = wv >> 2;     // 0..1 : pair of 16-row tiles
    const int half = lane >> 4;
    const int l16 = lane & 15;

    // gathered token rows for this wave's two A fragments
    const unsigned short* arow[2];
#pragma unroll
    for (int s = 0; s < 2; ++s) {
        int slot = m0 + (mg * 2 + s) * 16 + l16;
        int cl = slot < cnt ? slot : cnt - 1;
        int entry = list[e * T_TOK + cl];
        arow[s] = xb + (size_t)(entry >> 1) * D_DIM;
    }

    const unsigned short* grow =
        gwb + ((size_t)e * F_DIM + (f0 + nj * 16 + l16)) * D_DIM;
    const unsigned short* urow =
        uwb + ((size_t)e * F_DIM + (f0 + nj * 16 + l16)) * D_DIM;

    v8f accG[2] = {{}, {}}, accU[2] = {{}, {}};
    for (int k = 0; k < D_DIM; k += 32) {
        __builtin_prefetch(grow + k + 128, 0, 1);
        __builtin_prefetch(urow + k + 128, 0, 1);
        v16bf a0 = frag_bf16_rows(arow[0], k, lane);
        v16bf a1 = frag_bf16_rows(arow[1], k, lane);
        v16bf bg = frag_bf16_rows(grow, k, lane);
        v16bf bu = frag_bf16_rows(urow, k, lane);
        accG[0] = __builtin_amdgcn_wmma_f32_16x16x32_bf16(false, a0, false, bg,
                                                          (short)0, accG[0], false, false);
        accU[0] = __builtin_amdgcn_wmma_f32_16x16x32_bf16(false, a0, false, bu,
                                                          (short)0, accU[0], false, false);
        accG[1] = __builtin_amdgcn_wmma_f32_16x16x32_bf16(false, a1, false, bg,
                                                          (short)0, accG[1], false, false);
        accU[1] = __builtin_amdgcn_wmma_f32_16x16x32_bf16(false, a1, false, bu,
                                                          (short)0, accU[1], false, false);
    }

    // C layout: VGPR i -> row (i + half*8), col = l16
    const int col = f0 + nj * 16 + l16;
#pragma unroll
    for (int s = 0; s < 2; ++s) {
#pragma unroll
        for (int i = 0; i < 8; ++i) {
            int m = m0 + (mg * 2 + s) * 16 + i + half * 8;
            if (m < cnt) {
                int en = list[e * T_TOK + m];
                float g = accG[s][i], u = accU[s][i];
                // silu(g) * u via fast reciprocal (one v_rcp_f32)
                float h = g * u * __builtin_amdgcn_rcpf(1.0f + __expf(-g));
                hbuf[(size_t)en * F_DIM + col] = bf16_1(h);
            }
        }
    }
}

// ---------------- kernel 4: grouped down GEMM + weighted scatter ----------------
// grid = (D/64, T/64, E); block = 256 (8 waves); same wave mapping as k_gateup.
__global__ __launch_bounds__(256) void k_down(const unsigned short* __restrict__ hbuf,
                                              const unsigned short* __restrict__ dwb,
                                              const int* __restrict__ counts,
                                              const int* __restrict__ list,
                                              const float* __restrict__ wlist,
                                              float* __restrict__ out) {
    const int e = blockIdx.z;
    const int cnt = counts[e];
    const int m0 = blockIdx.y * 64;
    if (m0 >= cnt) return;
    const int o0 = blockIdx.x * 64;
    const int lane = threadIdx.x & 31;
    const int wv = threadIdx.x >> 5;
    const int nj = wv & 3;
    const int mg = wv >> 2;
    const int half = lane >> 4;
    const int l16 = lane & 15;

    const unsigned short* arow[2];
#pragma unroll
    for (int s = 0; s < 2; ++s) {
        int slot = m0 + (mg * 2 + s) * 16 + l16;
        int cl = slot < cnt ? slot : cnt - 1;
        int entry = list[e * T_TOK + cl];
        arow[s] = hbuf + (size_t)entry * F_DIM;
    }

    const unsigned short* brow =
        dwb + ((size_t)e * D_DIM + (o0 + nj * 16 + l16)) * F_DIM;

    v8f acc[2] = {{}, {}};
    for (int k = 0; k < F_DIM; k += 32) {
        __builtin_prefetch(brow + k + 128, 0, 1);
        v16bf a0 = frag_bf16_rows(arow[0], k, lane);
        v16bf a1 = frag_bf16_rows(arow[1], k, lane);
        v16bf b  = frag_bf16_rows(brow, k, lane);
        acc[0] = __builtin_amdgcn_wmma_f32_16x16x32_bf16(false, a0, false, b,
                                                         (short)0, acc[0], false, false);
        acc[1] = __builtin_amdgcn_wmma_f32_16x16x32_bf16(false, a1, false, b,
                                                         (short)0, acc[1], false, false);
    }

    const int col = o0 + nj * 16 + l16;
#pragma unroll
    for (int s = 0; s < 2; ++s) {
#pragma unroll
        for (int i = 0; i < 8; ++i) {
            int m = m0 + (mg * 2 + s) * 16 + i + half * 8;
            if (m < cnt) {
                int en = list[e * T_TOK + m];
                float w = wlist[e * T_TOK + m];
                int t = en >> 1;
                __hip_atomic_fetch_add(&out[(size_t)t * D_DIM + col], acc[s][i] * w,
                                       __ATOMIC_RELAXED, __HIP_MEMORY_SCOPE_AGENT);
            }
        }
    }
}

// ---------------- host side ----------------
extern "C" void kernel_launch(void* const* d_in, const int* in_sizes, int n_in,
                              void* d_out, int out_size, void* d_ws, size_t ws_size,
                              hipStream_t stream) {
    const float* x      = (const float*)d_in[0];   // [T, D]
    const float* rw     = (const float*)d_in[1];   // [E, D]
    const float* gate_w = (const float*)d_in[2];   // [E, F, D]
    const float* up_w   = (const float*)d_in[3];   // [E, F, D]
    const float* down_w = (const float*)d_in[4];   // [E, D, F]
    float* out = (float*)d_out;                    // [T, D]
    (void)in_sizes; (void)n_in; (void)out_size; (void)ws_size;

    // workspace layout (bf16 copies + routing metadata), ~136 MB total
    char* ws = (char*)d_ws;
    const size_t x_elems = (size_t)T_TOK * D_DIM;          // 4 Mi
    const size_t w_elems = (size_t)E_NUM * F_DIM * D_DIM;  // 16 Mi per weight
    const size_t h_elems = (size_t)2 * T_TOK * F_DIM;      // 16 Mi

    unsigned short* xb  = (unsigned short*)ws;
    unsigned short* gwb = xb + x_elems;
    unsigned short* uwb = gwb + w_elems;
    unsigned short* dwb = uwb + w_elems;
    unsigned short* hbuf = dwb + w_elems;
    int*   counts = (int*)(hbuf + h_elems);
    int*   list   = counts + 64;                           // 256B pad
    float* wlist  = (float*)(list + (size_t)E_NUM * T_TOK);

    // 0: zero output + counters
    k_zero<<<(T_TOK * D_DIM) / (256 * 4), 256, 0, stream>>>(out, counts);
    // 1: bulk fp32 -> bf16 conversions (memory-bound, ~300MB total traffic)
    k_cvt<<<x_elems / (256 * 4), 256, 0, stream>>>(x, xb);
    k_cvt<<<w_elems / (256 * 4), 256, 0, stream>>>(gate_w, gwb);
    k_cvt<<<w_elems / (256 * 4), 256, 0, stream>>>(up_w, uwb);
    k_cvt<<<w_elems / (256 * 4), 256, 0, stream>>>(down_w, dwb);
    // 2: router + compaction
    k_router<<<T_TOK / 256, 256, 0, stream>>>(x, rw, counts, list, wlist);
    // 3: gate/up grouped GEMM + SiLU
    k_gateup<<<dim3(F_DIM / 64, T_TOK / 64, E_NUM), 256, 0, stream>>>(
        xb, gwb, uwb, counts, list, hbuf);
    // 4: down grouped GEMM + weighted scatter-add
    k_down<<<dim3(D_DIM / 64, T_TOK / 64, E_NUM), 256, 0, stream>>>(
        hbuf, dwb, counts, list, wlist, out);
}
